// SparseMoE_80479097192507
// MI455X (gfx1250) — compile-verified, matrix operand order checked by reading, not run on hardware
//
#include <hip/hip_runtime.h>

#define N_TOK 65536
#define DIN   256
#define DOUT  256
#define NEXP  8

typedef __attribute__((ext_vector_type(16))) _Float16 v16h;
typedef __attribute__((ext_vector_type(8)))  _Float16 v8h;
typedef __attribute__((ext_vector_type(4)))  _Float16 v4h;
typedef __attribute__((ext_vector_type(8)))  float    v8f;
typedef __attribute__((ext_vector_type(4)))  float    v4f;

union V16U { v16h v; v8h h[2]; };

struct Pair { int s0, s1; float w0, w1; };

// ---------------------------------------------------------------------------
// Kernel 1: We[e][i][o] (fp32) -> BT[e][o][i] (f16).  B fragments then load
// with the exact same K-contiguous pattern as A fragments (ISA 7.12.2).
// ---------------------------------------------------------------------------
__global__ __launch_bounds__(256) void moe_prepw(const float* __restrict__ We,
                                                 _Float16* __restrict__ BT) {
  int id  = blockIdx.x * 256 + threadIdx.x;      // NEXP*DIN*DOUT threads
  int e   = id >> 16;
  int rem = id & 0xFFFF;
  int i   = rem >> 8;       // DIN index
  int o   = rem & 255;      // DOUT index (contiguous -> coalesced We reads)
  BT[(size_t)e * DIN * DOUT + (size_t)o * DIN + i] =
      (_Float16)We[(size_t)e * DIN * DOUT + (size_t)i * DOUT + o];
}

// ---------------------------------------------------------------------------
// Kernel 2: router.  One thread per token: logits over 8 experts, top-2 with
// earliest-index tie-break (matches lax.top_k), renormalized weights
// (softmax denominator cancels), f16 copy of x, grouped expert lists.
// ---------------------------------------------------------------------------
__global__ __launch_bounds__(256) void moe_router(
    const float* __restrict__ x, const float* __restrict__ Wr,
    _Float16* __restrict__ xh, int* __restrict__ counts,
    int* __restrict__ idxb, float* __restrict__ wtb,
    Pair* __restrict__ pairs) {
  __shared__ float sWr[DIN * NEXP];
  for (int i = threadIdx.x; i < DIN * NEXP; i += 256) sWr[i] = Wr[i];
  __syncthreads();

  int n = blockIdx.x * 256 + threadIdx.x;
  const v4f* xr = (const v4f*)(x + (size_t)n * DIN);
  v4h*       xo = (v4h*)(xh + (size_t)n * DIN);

  float acc[NEXP];
#pragma unroll
  for (int e = 0; e < NEXP; ++e) acc[e] = 0.0f;

  for (int i = 0; i < DIN / 4; ++i) {
    v4f xx = xr[i];
    v4h hh;
    hh.x = (_Float16)xx.x; hh.y = (_Float16)xx.y;
    hh.z = (_Float16)xx.z; hh.w = (_Float16)xx.w;
    xo[i] = hh;
    const float* w = &sWr[(i * 4) * NEXP];
#pragma unroll
    for (int e = 0; e < NEXP; ++e)
      acc[e] += xx.x * w[e] + xx.y * w[NEXP + e] +
                xx.z * w[2 * NEXP + e] + xx.w * w[3 * NEXP + e];
  }

  float b0 = -1e30f, b1 = -1e30f; int s0 = 0, s1 = 0;
#pragma unroll
  for (int e = 0; e < NEXP; ++e) {
    float l = acc[e];
    if (l > b0)      { b1 = b0; s1 = s0; b0 = l; s0 = e; }
    else if (l > b1) { b1 = l; s1 = e; }
  }
  float r   = __expf(b1 - b0);        // <= 1, numerically safe
  float inv = 1.0f / (1.0f + r);
  float w0n = inv, w1n = r * inv;

  int p0 = atomicAdd(&counts[s0], 1);
  idxb[(size_t)s0 * N_TOK + p0] = n;  wtb[(size_t)s0 * N_TOK + p0] = w0n;
  int p1 = atomicAdd(&counts[s1], 1);
  idxb[(size_t)s1 * N_TOK + p1] = n;  wtb[(size_t)s1 * N_TOK + p1] = w1n;

  Pair p; p.s0 = s0; p.s1 = s1; p.w0 = w0n; p.w1 = w1n;
  pairs[n] = p;
}

// ---------------------------------------------------------------------------
// Kernel 3: out[n][o] = w0*be[s0][o] + w1*be[s1][o]  (coalesced init)
// ---------------------------------------------------------------------------
__global__ __launch_bounds__(256) void moe_bias(const float* __restrict__ be,
                                                const Pair* __restrict__ pairs,
                                                float* __restrict__ out) {
  int n = blockIdx.x, o = threadIdx.x;
  Pair p = pairs[n];
  out[(size_t)n * DOUT + o] =
      p.w0 * be[(size_t)p.s0 * DOUT + o] + p.w1 * be[(size_t)p.s1 * DOUT + o];
}

// ---------------------------------------------------------------------------
// Kernel 4: grouped GEMM per expert with v_wmma_f32_16x16x32_f16.
// Block = 8 waves, each wave owns 16 gathered token rows (128 rows/block).
// BT[e] staged in LDS (row stride DIN+8 halves -> conflict-free b128 reads).
// ---------------------------------------------------------------------------
#define LPAD    8
#define LSTRIDE (DIN + LPAD)

__global__ __launch_bounds__(256) void moe_gemm(
    const _Float16* __restrict__ xh, const _Float16* __restrict__ BT,
    const int* __restrict__ counts, const int* __restrict__ idxb,
    const float* __restrict__ wtb, float* __restrict__ out) {
  extern __shared__ _Float16 lds[];

  const int e      = blockIdx.y;
  const int count  = counts[e];
  const int m_base = blockIdx.x * 128;
  if (m_base >= count) return;

  // Stage BT[e] (DOUT rows x DIN halves) into padded LDS, coalesced b128s.
  {
    const v8h* src = (const v8h*)(BT + (size_t)e * DIN * DOUT);
    for (int c = threadIdx.x; c < DOUT * (DIN / 8); c += 256) {
      int n8 = c >> 5;            // row (output col), 32 chunks per row
      int k8 = (c & 31) * 8;      // k offset in halves
      *(v8h*)(&lds[n8 * LSTRIDE + k8]) = src[c];
    }
  }
  __syncthreads();

  const int lane   = threadIdx.x & 31;
  const int wave   = threadIdx.x >> 5;
  const int laneHi = lane >> 4;          // 0: K 0..7/16..23, 1: K 8..15/24..31
  const int m_tile = m_base + wave * 16;

  // ---- A fragments: lane holds row M = lane&15 of the gathered tile ------
  int slotA = m_tile + (lane & 15);
  int tokA  = idxb[(size_t)e * N_TOK + (slotA < count ? slotA : m_base)];
  const _Float16* arow = xh + (size_t)tokA * DIN;
  V16U a[8];
#pragma unroll
  for (int kt = 0; kt < 8; ++kt) {
    const _Float16* p = arow + kt * 32 + laneHi * 8;
    a[kt].h[0] = *(const v8h*)(p);
    a[kt].h[1] = *(const v8h*)(p + 16);
  }

  // ---- store-side metadata: C/D VGPR r holds row M = r + 8*laneHi --------
  int   toks[8];
  float wrow[8];
#pragma unroll
  for (int rr = 0; rr < 8; ++rr) {
    int slot = m_tile + laneHi * 8 + rr;
    bool ok  = slot < count;
    toks[rr] = ok ? idxb[(size_t)e * N_TOK + slot] : -1;
    wrow[rr] = ok ? wtb[(size_t)e * N_TOK + slot] : 0.0f;
  }

  for (int nt = 0; nt < 16; ++nt) {
    v8f acc = {};
    const _Float16* brow = &lds[(nt * 16 + (lane & 15)) * LSTRIDE + laneHi * 8];
#pragma unroll
    for (int kt = 0; kt < 8; ++kt) {
      V16U b;
      b.h[0] = *(const v8h*)(brow + kt * 32);
      b.h[1] = *(const v8h*)(brow + kt * 32 + 16);
      acc = __builtin_amdgcn_wmma_f32_16x16x32_f16(
          false, a[kt].v, false, b.v, (short)0, acc, false, false);
    }
#pragma unroll
    for (int rr = 0; rr < 8; ++rr) {
      if (toks[rr] >= 0)
        unsafeAtomicAdd(&out[(size_t)toks[rr] * DOUT + nt * 16 + (lane & 15)],
                        acc[rr] * wrow[rr]);
    }
  }
}

// ---------------------------------------------------------------------------
extern "C" void kernel_launch(void* const* d_in, const int* in_sizes, int n_in,
                              void* d_out, int out_size, void* d_ws,
                              size_t ws_size, hipStream_t stream) {
  const float* x  = (const float*)d_in[0];
  const float* Wr = (const float*)d_in[1];
  const float* We = (const float*)d_in[2];
  const float* be = (const float*)d_in[3];
  float* out = (float*)d_out;

  char*  ws  = (char*)d_ws;
  size_t off = 0;
  auto alloc = [&](size_t bytes) {
    void* p = ws + off;
    off += (bytes + 255) & ~(size_t)255;
    return p;
  };
  _Float16* xh     = (_Float16*)alloc((size_t)N_TOK * DIN * 2);
  _Float16* BT     = (_Float16*)alloc((size_t)NEXP * DIN * DOUT * 2);
  int*      counts = (int*)alloc(NEXP * sizeof(int));
  int*      idxb   = (int*)alloc((size_t)NEXP * N_TOK * sizeof(int));
  float*    wtb    = (float*)alloc((size_t)NEXP * N_TOK * sizeof(float));
  Pair*     pairs  = (Pair*)alloc((size_t)N_TOK * sizeof(Pair));

  hipMemsetAsync(counts, 0, NEXP * sizeof(int), stream);
  moe_prepw<<<(NEXP * DIN * DOUT) / 256, 256, 0, stream>>>(We, BT);
  moe_router<<<N_TOK / 256, 256, 0, stream>>>(x, Wr, xh, counts, idxb, wtb,
                                              pairs);
  moe_bias<<<N_TOK, DOUT, 0, stream>>>(be, pairs, out);
  dim3 grid(N_TOK / 128, NEXP);
  moe_gemm<<<grid, 256, (size_t)DOUT * LSTRIDE * 2, stream>>>(
      xh, BT, counts, idxb, wtb, out);
}
